// FourierLayer_6889127543457
// MI455X (gfx1250) — matrix operand order repaired and checked by reading, not compiled, not run on hardware
//
#include <hip/hip_runtime.h>
#include <math.h>

// -----------------------------------------------------------------------------
// FNO spectral layer for MI455X (gfx1250, wave32, WMMA).
// All f16 WMMA operands (tables + intermediates) are stored in per-lane
// FRAGMENT layout: one 16x32 A-tile / 32x16 B-tile = 32 lanes x 16 halves
// (1KB), so each operand load is a single 32B vector load (2x b128).
// Pipeline:
//   k_init : DFT basis tables (fragment-packed f16, ortho 1/16 folded in)
//   k_s1   : Y = x @ Fw            (per (b,c): M=256,N=16,K=256, cos+sin)
//   k_s2   : Z = Fh @ Y            (complex, M=32,N=16,K=256)
//   k_s3   : Out = Z * W(mode)     (complex channel mix, M=8(b),N=64(o),K=64(i))
//   k_s4   : T = Ih @ Out          (complex, M=256,N=16,K=32)
//   k_s5   : out = Re-iW(T) + conv_w@x + b   (fused, coalesced b128 writes)
// -----------------------------------------------------------------------------

typedef __attribute__((ext_vector_type(16))) _Float16 v16h;
typedef __attribute__((ext_vector_type(8)))  float    v8f;

#define WSCALE 16.0f  // spectral weights scaled x16 into f16 range; /16 folded into iw

__device__ __forceinline__ int lane() { return threadIdx.x & 31; }

__device__ __forceinline__ v8f wmma_f16(v16h a, v16h b, v8f c) {
  return __builtin_amdgcn_wmma_f32_16x16x32_f16(false, a, false, b, (short)0, c,
                                                false, false);
}

// ---- fragment layout helpers (ISA 7.12.2 layouts) ---------------------------
// A (16x32): element (m,k) -> lane = ((k&15)>>3)*16 + m, slot = (k&7) + (k&16 ? 8 : 0)
__device__ __forceinline__ int frag_a_idx(int k, int m) {
  int hi = (k & 15) >> 3;
  int slot = (k & 7) + ((k & 16) ? 8 : 0);
  return (hi * 16 + m) * 16 + slot;
}
// B (32x16): element (k,n) -> lane = ((k&15)>>3)*16 + n, slot = (k&7) + (k&16 ? 8 : 0)
__device__ __forceinline__ int frag_b_idx(int k, int n) {
  return (((k & 15) >> 3) * 16 + n) * 16 + (k & 7) + ((k & 16) ? 8 : 0);
}
// Load one fragment tile (32 lanes x 16 halves): single 32B vector load
__device__ __forceinline__ v16h load_frag(const _Float16* __restrict__ buf, int tile) {
  return *(const v16h*)(buf + (size_t)tile * 512 + (size_t)lane() * 16);
}

// A-matrix K index for half-slot t and lane-half hi (forward mapping)
__device__ __forceinline__ int a_k(int t, int hi) {
  return (t < 8) ? (hi * 8 + t) : (8 + hi * 8 + t);
}

// Load 16x32 A tile from row-major f32 (x input): 8+8 contiguous floats/lane -> b128s
__device__ __forceinline__ v16h load_A_f32(const float* src, int stride, int row0, int col0) {
  int l = lane(), m = l & 15, hi = l >> 4;
  const float* p = src + (size_t)(row0 + m) * stride + col0;
  v16h a;
#pragma unroll
  for (int t = 0; t < 16; ++t) a[t] = (_Float16)p[a_k(t, hi)];
  return a;
}

// ---------------------------------------------------------------- k_init -----
__global__ void k_init(const float* __restrict__ conv_w,
                       _Float16* fwc, _Float16* fws,
                       _Float16* fhc, _Float16* fhs, _Float16* fhsn,
                       _Float16* ihc, _Float16* ihs, _Float16* ihsn,
                       _Float16* iw, _Float16* cw16) {
  const float TP = 6.283185307179586f / 256.0f;
  int tid = blockIdx.x * blockDim.x + threadIdx.x;
  int nth = gridDim.x * blockDim.x;
  // fw: B-fragments, tile = w>>5; element (k=w&31, n=ky)
  for (int idx = tid; idx < 256 * 16; idx += nth) {
    int w = idx >> 4, ky = idx & 15;
    float th = TP * (float)(w * ky);
    int a = (w >> 5) * 512 + frag_b_idx(w & 31, ky);
    fwc[a] = (_Float16)(cosf(th) * (1.0f / 16.0f));
    fws[a] = (_Float16)(sinf(th) * (1.0f / 16.0f));
  }
  // fh: A-fragments, tile = (kx>>4)*8 + (h>>5); element (m=kx&15, k=h&31)
  for (int idx = tid; idx < 32 * 256; idx += nth) {
    int kx = idx >> 8, h = idx & 255;
    int f = (kx < 16) ? kx : (kx - 32);
    float th = TP * (float)(f * h);
    float c = cosf(th) * (1.0f / 16.0f), s = sinf(th) * (1.0f / 16.0f);
    int a = ((kx >> 4) * 8 + (h >> 5)) * 512 + frag_a_idx(h & 31, kx & 15);
    fhc[a] = (_Float16)c; fhs[a] = (_Float16)s; fhsn[a] = (_Float16)(-s);
  }
  // ih: A-fragments, tile = h>>4; element (m=h&15, k=kx)
  for (int idx = tid; idx < 256 * 32; idx += nth) {
    int h = idx >> 5, kx = idx & 31;
    int f = (kx < 16) ? kx : (kx - 32);
    float th = TP * (float)(f * h);
    float c = cosf(th) * (1.0f / 16.0f), s = sinf(th) * (1.0f / 16.0f);
    int a = (h >> 4) * 512 + frag_a_idx(kx, h & 15);
    ihc[a] = (_Float16)c; ihs[a] = (_Float16)s; ihsn[a] = (_Float16)(-s);
  }
  // iw: B-fragments, tile = w>>4; element (k, n=w&15)
  // rows 0..15: c_ky*cos ; rows 16..31: -2*sin ; /16 (ortho) * /16 (WSCALE comp)
  for (int idx = tid; idx < 32 * 256; idx += nth) {
    int k = idx >> 8, w = idx & 255;
    float v;
    if (k < 16) {
      float cc = (k == 0) ? 1.0f : 2.0f;
      v = cc * cosf(TP * (float)(k * w)) * (1.0f / 256.0f);
    } else {
      int ky = k - 16;
      v = (ky == 0) ? 0.0f : (-2.0f * sinf(TP * (float)(ky * w)) * (1.0f / 256.0f));
    }
    iw[(w >> 4) * 512 + frag_b_idx(k, w & 15)] = (_Float16)v;
  }
  // cw16: B-fragments, tile = (i>>5)*4 + (o>>4); element (k=i&31, n=o&15)
  for (int idx = tid; idx < 64 * 64; idx += nth) {
    int i = idx >> 6, o = idx & 63;
    cw16[((i >> 5) * 4 + (o >> 4)) * 512 + frag_b_idx(i & 31, o & 15)] =
        (_Float16)conv_w[o * 64 + i];
  }
}

// ------------------------------------------------------------------ k_s1 -----
// Y[b,c][h][ky] = sum_w x * e^{-i th}/16; store as B-fragments for k_s2
__global__ __launch_bounds__(256) void k_s1(const float* __restrict__ x,
                                            const _Float16* __restrict__ fwc,
                                            const _Float16* __restrict__ fws,
                                            _Float16* yr, _Float16* yi) {
  int bc = blockIdx.x;               // b*64+c
  int wave = threadIdx.x >> 5;
  int l = lane(), n = l & 15, hi = l >> 4;
  const float* xb = x + (size_t)bc * (256 * 256);
  _Float16* yrb = yr + (size_t)bc * 4096;   // 8 tiles of 512 halves
  _Float16* yib = yi + (size_t)bc * 4096;
#pragma unroll
  for (int j = 0; j < 2; ++j) {
    int ht = wave * 2 + j;
    v8f ac = {}, as = {};
#pragma unroll
    for (int k = 0; k < 8; ++k) {
      v16h A  = load_A_f32(xb, 256, ht * 16, k * 32);
      v16h Bc = load_frag(fwc, k);
      v16h Bs = load_frag(fws, k);
      ac = wmma_f16(A, Bc, ac);
      as = wmma_f16(A, Bs, as);
    }
    // scatter D (h, ky=n) into y B-fragment (tile = h>>5, k = h&31)
#pragma unroll
    for (int v = 0; v < 8; ++v) {
      int h = ht * 16 + v + hi * 8;
      int a = (h >> 5) * 512 + frag_b_idx(h & 31, n);
      yrb[a] = (_Float16)ac[v];
      yib[a] = (_Float16)(-as[v]);
    }
  }
}

// ------------------------------------------------------------------ k_s2 -----
// Zr = fhc@Yr + fhs@Yi ; Zi = fhc@Yi + (-fhs)@Yr ; store as A-fragments for k_s3
__global__ __launch_bounds__(256) void k_s2(const _Float16* __restrict__ yr,
                                            const _Float16* __restrict__ yi,
                                            const _Float16* __restrict__ fhc,
                                            const _Float16* __restrict__ fhs,
                                            const _Float16* __restrict__ fhsn,
                                            _Float16* zr, _Float16* zi) {
  int bc = blockIdx.x;
  int wave = threadIdx.x >> 5;
  if (wave >= 4) return;             // 2 m-tiles x {re,im}
  int mt = wave >> 1, isI = wave & 1;
  int l = lane(), n = l & 15, hi = l >> 4;
  const _Float16* y1 = (isI ? yi : yr) + (size_t)bc * 4096;
  const _Float16* y2 = (isI ? yr : yi) + (size_t)bc * 4096;
  const _Float16* a2 = isI ? fhsn : fhs;
  v8f acc = {};
#pragma unroll
  for (int k = 0; k < 8; ++k) {
    v16h A1 = load_frag(fhc, mt * 8 + k);
    v16h B1 = load_frag(y1, k);
    acc = wmma_f16(A1, B1, acc);
    v16h A2 = load_frag(a2, mt * 8 + k);
    v16h B2 = load_frag(y2, k);
    acc = wmma_f16(A2, B2, acc);
  }
  // scatter D (kx, ky=n) value of (b,i)=(bc>>6, bc&63) into z A-fragment
  // z layout: [mode][ks][512]
  {
    int b = bc >> 6, c = bc & 63;
    int ks = c >> 5;
    int ai = frag_a_idx(c & 31, b);
    _Float16* dst = isI ? zi : zr;
#pragma unroll
    for (int v = 0; v < 8; ++v) {
      int kx = mt * 16 + v + hi * 8;
      int mode = kx * 16 + n;
      dst[(size_t)(mode * 2 + ks) * 512 + ai] = (_Float16)acc[v];
    }
  }
}

// ------------------------------------------------------------------ k_s3 -----
// B: weight tile [32i x 16o], w layout [i][o][16][16] f32, scaled by sgn
__device__ __forceinline__ v16h load_B_w(const float* w, int i0, int o0, int moff, float sgn) {
  int l = lane(), n = l & 15, hi = l >> 4;
  v16h b;
#pragma unroll
  for (int t = 0; t < 16; ++t) {
    int i = i0 + a_k(t, hi);
    b[t] = (_Float16)(sgn * w[((size_t)i * 64 + (o0 + n)) * 256 + moff]);
  }
  return b;
}
// Or = Zr@Wr - Zi@Wi ; Oi = Zr@Wi + Zi@Wr ; store as B-fragments for k_s4
__global__ __launch_bounds__(256) void k_s3(const _Float16* __restrict__ zr,
                                            const _Float16* __restrict__ zi,
                                            const float* __restrict__ w1r,
                                            const float* __restrict__ w1i,
                                            const float* __restrict__ w2r,
                                            const float* __restrict__ w2i,
                                            _Float16* orr, _Float16* oi) {
  int mode = blockIdx.x;             // kx*16 + ky
  int kx = mode >> 4, ky = mode & 15;
  int wave = threadIdx.x >> 5;
  int nt = wave >> 1, isI = wave & 1;          // 4 o-tiles x {re,im}
  int l = lane(), m = l & 15, hi = l >> 4;
  const float* wr = (kx < 16) ? w1r : w2r;
  const float* wi = (kx < 16) ? w1i : w2i;
  int m1 = (kx < 16) ? kx : (kx - 16);
  int moff = m1 * 16 + ky;
  v8f acc = {};
#pragma unroll
  for (int ks = 0; ks < 2; ++ks) {
    v16h A1, A2;
    if (m < 8) {                     // b rows 0..7 only; rest of fragment is zero
      A1 = load_frag(zr, mode * 2 + ks);
      A2 = load_frag(zi, mode * 2 + ks);
    } else {
#pragma unroll
      for (int t = 0; t < 16; ++t) { A1[t] = (_Float16)0.0f; A2[t] = (_Float16)0.0f; }
    }
    v16h B1 = load_B_w(isI ? wi : wr, ks * 32, nt * 16, moff, WSCALE);
    v16h B2 = load_B_w(isI ? wr : wi, ks * 32, nt * 16, moff, isI ? WSCALE : -WSCALE);
    acc = wmma_f16(A1, B1, acc);
    acc = wmma_f16(A2, B2, acc);
  }
  // scatter D rows b<8 into o B-fragment: o layout [bo][512], element (k=kx, n=ky)
  {
    _Float16* dst = isI ? oi : orr;
    if (hi == 0) {
      int bi = frag_b_idx(kx, ky);
#pragma unroll
      for (int v = 0; v < 8; ++v)
        dst[(size_t)(v * 64 + nt * 16 + m) * 512 + bi] = (_Float16)acc[v];
    }
  }
}

// ------------------------------------------------------------------ k_s4 -----
// Tr = ihc@Or + (-ihs)@Oi ; Ti = ihc@Oi + ihs@Or
// store Tr/Ti merged into one A-fragment buffer for k_s5: K 0..15 = Tr[ky], 16..31 = Ti[ky]
__global__ __launch_bounds__(256) void k_s4(const _Float16* __restrict__ orr,
                                            const _Float16* __restrict__ oi,
                                            const _Float16* __restrict__ ihc,
                                            const _Float16* __restrict__ ihs,
                                            const _Float16* __restrict__ ihsn,
                                            _Float16* tfrag) {
  int bo = blockIdx.x;               // b*64 + o
  int wave = threadIdx.x >> 5;
  int l = lane(), n = l & 15, hi = l >> 4;
#pragma unroll
  for (int t4 = 0; t4 < 4; ++t4) {
    int task = wave * 4 + t4;        // 16 h-tiles x {re,im}
    int ht = task >> 1, isI = task & 1;
    v16h A1 = load_frag(ihc, ht);
    v16h B1 = load_frag(isI ? oi : orr, bo);
    v16h A2 = load_frag(isI ? ihs : ihsn, ht);
    v16h B2 = load_frag(isI ? orr : oi, bo);
    v8f acc = {};
    acc = wmma_f16(A1, B1, acc);
    acc = wmma_f16(A2, B2, acc);
    // scatter D (h, ky=n) into t A-fragment: tile = bo*16+ht, k = n (+16 if imag)
    int ai = frag_a_idx(isI ? (16 + n) : n, 0);
#pragma unroll
    for (int v = 0; v < 8; ++v) {
      int m = v + hi * 8;            // h & 15
      tfrag[(size_t)(bo * 16 + ht) * 512 + ai + m * 16] = (_Float16)acc[v];
    }
  }
}

// ------------------------------------------------------------------ k_s5 -----
// out[b,o,h0:16,w0:16] = T_frag @ iw  +  conv_w@x + b, staged in swizzled 64KB LDS.
__global__ __launch_bounds__(256) void k_s5(const float* __restrict__ x,
                                            const _Float16* __restrict__ tfrag,
                                            const _Float16* __restrict__ iw,
                                            const _Float16* __restrict__ cw16,
                                            const float* __restrict__ conv_b,
                                            float* __restrict__ out) {
  __shared__ float out_s[256 * 64];  // [p=hh*16+ww][o], bank-swizzled
  int wt = blockIdx.x, htl = blockIdx.y, b = blockIdx.z;
  int h0 = htl * 16, w0 = wt * 16;
  int tid = threadIdx.x;
  int wave = tid >> 5, l = tid & 31;
  int m = l & 15, hi = l >> 4;
  auto sw = [](int p, int o) { return p * 64 + (o ^ ((p * 4) & 63)); };

  // warm L2/L0 for the conv branch's 64 scattered i-planes while spectral runs
#pragma unroll
  for (int r = 0; r < 2; ++r) {
    int i = r * 32 + l;
    const float* pf = &x[(((size_t)b * 64 + i) * 256 + (h0 + wave * 2)) * 256 + w0];
    __builtin_prefetch(pf, 0, 3);
    __builtin_prefetch(pf + 256, 0, 3);
  }

  // ---- spectral branch: per o, D[16h x 16w] = {Tr|Ti}[16x32] @ iw[32 x 16w]
  v16h Biw = load_frag(iw, wt);
#pragma unroll
  for (int j = 0; j < 8; ++j) {
    int o = wave * 8 + j;
    v16h A = load_frag(tfrag, (b * 64 + o) * 16 + htl);
    v8f d = {};
    d = wmma_f16(A, Biw, d);
#pragma unroll
    for (int v = 0; v < 8; ++v) out_s[sw((v + hi * 8) * 16 + m, o)] = d[v];
  }
  __syncthreads();

  // ---- conv branch: [p x o] += X[p x i] @ cw16[i x o], C init from spectral
  v16h Bw[2][4];
#pragma unroll
  for (int ks = 0; ks < 2; ++ks)
#pragma unroll
    for (int nt = 0; nt < 4; ++nt) Bw[ks][nt] = load_frag(cw16, ks * 4 + nt);

#pragma unroll
  for (int mi = 0; mi < 2; ++mi) {
    int mt = wave * 2 + mi;          // tile rows p = mt*16 + r  (hh = mt, ww = r)
    v8f acc[4];
#pragma unroll
    for (int nt = 0; nt < 4; ++nt) {
      int o = nt * 16 + m;
#pragma unroll
      for (int v = 0; v < 8; ++v) acc[nt][v] = out_s[sw(mt * 16 + v + hi * 8, o)];
    }
#pragma unroll
    for (int ks = 0; ks < 2; ++ks) {
      v16h A;
#pragma unroll
      for (int t = 0; t < 16; ++t) {
        int i = ks * 32 + a_k(t, hi);
        A[t] = (_Float16)x[(((size_t)b * 64 + i) * 256 + (h0 + mt)) * 256 + w0 + m];
      }
#pragma unroll
      for (int nt = 0; nt < 4; ++nt) acc[nt] = wmma_f16(A, Bw[ks][nt], acc[nt]);
    }
#pragma unroll
    for (int nt = 0; nt < 4; ++nt) {
      int o = nt * 16 + m;
      float bias = conv_b[o];
#pragma unroll
      for (int v = 0; v < 8; ++v)
        out_s[sw(mt * 16 + v + hi * 8, o)] = acc[nt][v] + bias;
    }
  }
  __syncthreads();

  // ---- coalesced float4 writes
#pragma unroll
  for (int r = 0; r < 4; ++r) {
    int pair = tid + 256 * r;        // 1024 (o,hh) pairs
    int o = pair >> 4, hh = pair & 15;
    float tmp[16];
#pragma unroll
    for (int ww = 0; ww < 16; ++ww) tmp[ww] = out_s[sw(hh * 16 + ww, o)];
    float* dst = out + (((size_t)b * 64 + o) * 256 + (h0 + hh)) * 256 + w0;
#pragma unroll
    for (int q = 0; q < 4; ++q)
      ((float4*)dst)[q] = make_float4(tmp[4 * q], tmp[4 * q + 1], tmp[4 * q + 2], tmp[4 * q + 3]);
  }
}

// ------------------------------------------------------------------------------
extern "C" void kernel_launch(void* const* d_in, const int* in_sizes, int n_in,
                              void* d_out, int out_size, void* d_ws, size_t ws_size,
                              hipStream_t stream) {
  (void)in_sizes; (void)n_in; (void)out_size; (void)ws_size;
  const float* x      = (const float*)d_in[0];
  const float* w1r    = (const float*)d_in[1];
  const float* w1i    = (const float*)d_in[2];
  const float* w2r    = (const float*)d_in[3];
  const float* w2i    = (const float*)d_in[4];
  const float* conv_w = (const float*)d_in[5];
  const float* conv_b = (const float*)d_in[6];
  float* out = (float*)d_out;

  char* ws = (char*)d_ws;                       // needs ~19.3 MB
  _Float16* fwc  = (_Float16*)(ws + 0);         //  8 KB, 8 B-frag tiles
  _Float16* fws  = (_Float16*)(ws + 8192);      //  8 KB
  _Float16* fhc  = (_Float16*)(ws + 16384);     // 16 KB, 16 A-frag tiles
  _Float16* fhs  = (_Float16*)(ws + 32768);     // 16 KB
  _Float16* fhsn = (_Float16*)(ws + 49152);     // 16 KB
  _Float16* ihc  = (_Float16*)(ws + 65536);     // 16 KB, 16 A-frag tiles
  _Float16* ihs  = (_Float16*)(ws + 81920);     // 16 KB
  _Float16* ihsn = (_Float16*)(ws + 98304);     // 16 KB
  _Float16* iw   = (_Float16*)(ws + 114688);    // 16 KB, 16 B-frag tiles
  _Float16* cw16 = (_Float16*)(ws + 131072);    //  8 KB, 8 B-frag tiles
  size_t off = 262144;
  _Float16* yr    = (_Float16*)(ws + off); off += 4194304;  // [bc][8 tiles]
  _Float16* yi    = (_Float16*)(ws + off); off += 4194304;
  _Float16* zr    = (_Float16*)(ws + off); off += 1048576;  // [mode][2 tiles]
  _Float16* zi    = (_Float16*)(ws + off); off += 1048576;
  _Float16* orr   = (_Float16*)(ws + off); off += 524288;   // [bo][1 tile]
  _Float16* oi    = (_Float16*)(ws + off); off += 524288;
  _Float16* tfrag = (_Float16*)(ws + off); off += 8388608;  // [bo][16 tiles] Tr|Ti

  k_init<<<64, 256, 0, stream>>>(conv_w, fwc, fws, fhc, fhs, fhsn, ihc, ihs, ihsn, iw, cw16);
  k_s1<<<512, 256, 0, stream>>>(x, fwc, fws, yr, yi);
  k_s2<<<512, 256, 0, stream>>>(yr, yi, fhc, fhs, fhsn, zr, zi);
  k_s3<<<512, 256, 0, stream>>>(zr, zi, w1r, w1i, w2r, w2i, orr, oi);
  k_s4<<<512, 256, 0, stream>>>(orr, oi, ihc, ihs, ihsn, tfrag);
  k_s5<<<dim3(16, 16, 8), 256, 0, stream>>>(x, tfrag, iw, cw16, conv_b, out);
}
